// QLinearLayer_88459146428820
// MI455X (gfx1250) — matrix operand order repaired and checked by reading, not compiled, not run on hardware
//
#include <hip/hip_runtime.h>
#include <hip/hip_bf16.h>

// ---------------------------------------------------------------------------
// Types for CDNA5 WMMA (wave32). v16bf = A/B fragment, v8f = C/D fragment.
// ---------------------------------------------------------------------------
typedef __bf16 bf16_t;
typedef bf16_t v16bf __attribute__((ext_vector_type(16)));
typedef float  v8f   __attribute__((ext_vector_type(8)));
typedef float  v4f   __attribute__((ext_vector_type(4)));

// Problem shape (fixed by reference)
#define MDIM 8192
#define NDIM 4096
#define KDIM 4096

// GEMM tiling
#define BM 128
#define BN 128
#define BK 32

// Async global->LDS (gfx1250 GLOBAL_LOAD_ASYNC_TO_LDS_*, ASYNCcnt) if the
// toolchain exposes the builtins; otherwise fall back to load+ds_store.
#if defined(__has_builtin)
#  if __has_builtin(__builtin_amdgcn_global_load_async_to_lds_b128) && \
      __has_builtin(__builtin_amdgcn_s_wait_asynccnt)
#    define USE_ASYNC_COPY 1
#  endif
#endif
#ifndef USE_ASYNC_COPY
#  define USE_ASYNC_COPY 0
#endif

#if USE_ASYNC_COPY
// exact param type per hipcc diagnostic: vector_size(16) int, AS1 src / AS3 dst
typedef int v4i_ld __attribute__((vector_size(16)));
typedef __attribute__((address_space(1))) v4i_ld* g_v4i_ptr;
typedef __attribute__((address_space(3))) v4i_ld* l_v4i_ptr;
#endif

// ---------------------------------------------------------------------------
// Helpers
// ---------------------------------------------------------------------------
__device__ __forceinline__ float bf16rf(float v) {
    return __bfloat162float(__float2bfloat16(v));
}

// e2m1 nearest-value quantize, ties keep the SMALLER value (searchsorted left).
__device__ __forceinline__ float q_e2m1(float t) {
    float r = -6.0f;
    r = (t > -5.0f)  ? -4.0f : r;
    r = (t > -3.5f)  ? -3.0f : r;
    r = (t > -2.5f)  ? -2.0f : r;
    r = (t > -1.75f) ? -1.5f : r;
    r = (t > -1.25f) ? -1.0f : r;
    r = (t > -0.75f) ? -0.5f : r;
    r = (t > -0.25f) ?  0.0f : r;
    r = (t >  0.25f) ?  0.5f : r;
    r = (t >  0.75f) ?  1.0f : r;
    r = (t >  1.25f) ?  1.5f : r;
    r = (t >  1.75f) ?  2.0f : r;
    r = (t >  2.5f)  ?  3.0f : r;
    r = (t >  3.5f)  ?  4.0f : r;
    r = (t >  5.0f)  ?  6.0f : r;
    return r;
}

// ue4m3 quantize of a positive scale, clipped to [0.002, 448]
__device__ __forceinline__ float q_ue4m3(float t) {
    t = fminf(fmaxf(t, 0.002f), 448.0f);
    float e  = floorf(log2f(t + 1e-9f));
    float p  = exp2f(e);
    float m  = t / p - 1.0f;
    float qm = rintf(m * 8.0f) * 0.125f;
    return (1.0f + qm) * p;
}

// NVFP4 quantize of N values; writes bf16 to outq, and (optionally) the f32
// value of that bf16 back to outf (needed for the residual path).
template <int N>
__device__ __forceinline__ void nvfp4_quant(const float* in, bf16_t* outq, float* outf) {
    float ma = 0.0f;
#pragma unroll
    for (int i = 0; i < N; ++i) ma = fmaxf(ma, fabsf(in[i]));
    float scale = ma * (1.0f / 6.0f);
    scale = (scale == 0.0f) ? 1e-9f : scale;
    float dq = q_ue4m3(scale);
#pragma unroll
    for (int i = 0; i < N; ++i) {
        float q = q_e2m1(in[i] / dq);
        float v = bf16rf(q * dq);
        outq[i] = (bf16_t)v;
        if (outf) outf[i] = v;
    }
}

// ---------------------------------------------------------------------------
// Kernel 0: zero the atomic-max slots in the workspace header
// ---------------------------------------------------------------------------
__global__ void init_hdr_kernel(unsigned* hdr_u) {
    hdr_u[0] = 0u;
    hdr_u[1] = 0u;
}

// ---------------------------------------------------------------------------
// Kernel 1: max |bf16(v)| over a flat array -> atomicMax on float bits
// ---------------------------------------------------------------------------
__global__ __launch_bounds__(256) void maxabs_kernel(const float* __restrict__ in,
                                                     size_t n, unsigned* slot) {
    float m = 0.0f;
    for (size_t i = (size_t)blockIdx.x * blockDim.x + threadIdx.x; i < n;
         i += (size_t)gridDim.x * blockDim.x) {
        m = fmaxf(m, fabsf(bf16rf(in[i])));
    }
    __shared__ float red[256];
    red[threadIdx.x] = m;
    __syncthreads();
    for (int s = 128; s > 0; s >>= 1) {
        if ((int)threadIdx.x < s)
            red[threadIdx.x] = fmaxf(red[threadIdx.x], red[threadIdx.x + s]);
        __syncthreads();
    }
    if (threadIdx.x == 0) atomicMax(slot, __float_as_uint(red[0]));
}

// ---------------------------------------------------------------------------
// Kernel 2: finalize global scales
// ---------------------------------------------------------------------------
__global__ void scales_kernel(const unsigned* hdr_u, float* hdr_f) {
    float sx = fmaxf(__uint_as_float(hdr_u[0]) * (1.0f / (448.0f * 6.0f)), 1e-9f);
    float sw = fmaxf(__uint_as_float(hdr_u[1]) * (1.0f / (448.0f * 6.0f)), 1e-9f);
    hdr_f[2] = sx;
    hdr_f[3] = sw;
    hdr_f[4] = sx * sw;
}

// ---------------------------------------------------------------------------
// Kernel 3: weight quantization. One thread per group of 32 along K.
// ---------------------------------------------------------------------------
__global__ __launch_bounds__(256) void quantw_kernel(const float* __restrict__ w,
                                                     const float* __restrict__ hdr,
                                                     bf16_t* __restrict__ wq) {
    const float sw = hdr[3];
    size_t g = (size_t)blockIdx.x * blockDim.x + threadIdx.x;  // 524288 groups
    size_t base = g * 32;
    float v[32];
#pragma unroll
    for (int j = 0; j < 32; ++j) v[j] = bf16rf(w[base + j]) / sw;
    bf16_t q[32];
    nvfp4_quant<32>(v, q, nullptr);
#pragma unroll
    for (int j = 0; j < 32; ++j) wq[base + j] = q[j];
}

// ---------------------------------------------------------------------------
// Kernel 4: activation quantization (sparse top-2-of-4-pairs + residual)
// ---------------------------------------------------------------------------
__global__ __launch_bounds__(256) void quantx_kernel(const float* __restrict__ x,
                                                     const float* __restrict__ hdr,
                                                     bf16_t* __restrict__ xsq,
                                                     bf16_t* __restrict__ xrq) {
    const float sx = hdr[2];
    size_t g = (size_t)blockIdx.x * blockDim.x + threadIdx.x;  // 1048576 groups
    size_t base = g * 32;

    float xf[32];
#pragma unroll
    for (int j = 0; j < 32; ++j) xf[j] = bf16rf(x[base + j]) / sx;

    float xs[32];
#pragma unroll
    for (int s = 0; s < 4; ++s) {
        float sc[4];
#pragma unroll
        for (int p = 0; p < 4; ++p)
            sc[p] = fmaxf(fabsf(xf[s * 8 + 2 * p]), fabsf(xf[s * 8 + 2 * p + 1]));
        int i1 = 0;
#pragma unroll
        for (int p = 1; p < 4; ++p) i1 = (sc[p] > sc[i1]) ? p : i1;
        int i2 = -1;
#pragma unroll
        for (int p = 0; p < 4; ++p) {
            bool better = (i2 < 0) || (sc[p] > sc[i2]);
            i2 = (p != i1 && better) ? p : i2;
        }
#pragma unroll
        for (int p = 0; p < 4; ++p) {
            float keep = (p == i1 || p == i2) ? 1.0f : 0.0f;
            xs[s * 8 + 2 * p]     = xf[s * 8 + 2 * p] * keep;
            xs[s * 8 + 2 * p + 1] = xf[s * 8 + 2 * p + 1] * keep;
        }
    }

    bf16_t qs[32];
    float qsf[32];
    nvfp4_quant<32>(xs, qs, qsf);
#pragma unroll
    for (int j = 0; j < 32; ++j) xsq[base + j] = qs[j];

    float r[32];
#pragma unroll
    for (int j = 0; j < 32; ++j) r[j] = xf[j] - qsf[j];
    bf16_t qr[32];
    nvfp4_quant<16>(r, qr, nullptr);
    nvfp4_quant<16>(r + 16, qr + 16, nullptr);
#pragma unroll
    for (int j = 0; j < 32; ++j) xrq[base + j] = qr[j];
}

// ---------------------------------------------------------------------------
// Tile copy: 32 bytes per thread per array (2 x b128), async when available
// ---------------------------------------------------------------------------
__device__ __forceinline__ void copy16(const bf16_t* g, bf16_t* l) {
#if USE_ASYNC_COPY
    __builtin_amdgcn_global_load_async_to_lds_b128(
        (g_v4i_ptr)g, (l_v4i_ptr)l, 0, 0);
    __builtin_amdgcn_global_load_async_to_lds_b128(
        (g_v4i_ptr)(g + 8), (l_v4i_ptr)(l + 8), 0, 0);
#else
    ((v4f*)l)[0] = ((const v4f*)g)[0];
    ((v4f*)l)[1] = ((const v4f*)g)[1];
#endif
}

__device__ __forceinline__ void wait_copies() {
#if USE_ASYNC_COPY
    __builtin_amdgcn_s_wait_asynccnt(0);
#endif
}

// ---------------------------------------------------------------------------
// Kernel 5: fused dual-GEMM with double-buffered LDS pipeline
//   C = (Xs_q @ Wq^T + Xr_q @ Wq^T) * (sx*sw) + bias   (bf16 out)
//   128x128 block tile, BK=32, 8 waves (2x4), each wave 64x32 = 4x2 WMMA tiles.
//   Next k-tile streams global->LDS (async DMA when available) while the
//   current tile runs 16 v_wmma_f32_16x16x32_bf16 per wave.
// ---------------------------------------------------------------------------
__global__ __launch_bounds__(256) void gemm_kernel(const bf16_t* __restrict__ As_g,
                                                   const bf16_t* __restrict__ Ar_g,
                                                   const bf16_t* __restrict__ Wq,
                                                   const float* __restrict__ hdr,
                                                   const float* __restrict__ bias,
                                                   __hip_bfloat16* __restrict__ out) {
    __shared__ bf16_t As[2][BM * BK];
    __shared__ bf16_t Ar[2][BM * BK];
    __shared__ bf16_t Bs[2][BN * BK];

    const int tid   = threadIdx.x;
    const int lane  = tid & 31;
    const int wave  = tid >> 5;
    const int waveM = wave >> 2;   // 0..1 -> 64 rows each
    const int waveN = wave & 3;    // 0..3 -> 32 cols each
    const int m0    = blockIdx.y * BM;
    const int n0    = blockIdx.x * BN;

    const float scale = hdr[4];

    // cooperative copy: each thread moves 32B per array per k-step
    const int crow = tid >> 1;           // 0..127
    const int coff = (tid & 1) * 16;     // element offset {0,16}
    const size_t ga = (size_t)(m0 + crow) * KDIM + coff;
    const size_t gb = (size_t)(n0 + crow) * KDIM + coff;
    const int    lo = crow * BK + coff;

    // fragment addressing per CDNA5 16-bit layouts
    const int arow = lane & 15;          // A: M = lane%16
    const int ka   = (lane >> 4) * 8;    // A: K chunks [ka, ka+8) and [ka+16, ka+24)
    const int nrow = lane & 15;          // B: N = lane%16
    const int kb   = (lane >> 4) * 8;    // B: K chunks [kb, kb+8) and [kb+16, kb+24)

    v8f acc[4][2] = {};

    // prologue: fill buffer 0
    copy16(As_g + ga, &As[0][lo]);
    copy16(Ar_g + ga, &Ar[0][lo]);
    copy16(Wq   + gb, &Bs[0][lo]);
    wait_copies();
    __syncthreads();

    for (int k0 = 0; k0 < KDIM; k0 += BK) {
        const int buf  = (k0 / BK) & 1;
        const int nbuf = buf ^ 1;

        // kick off the next tile's global->LDS copies (overlaps with WMMA)
        if (k0 + BK < KDIM) {
            copy16(As_g + ga + k0 + BK, &As[nbuf][lo]);
            copy16(Ar_g + ga + k0 + BK, &Ar[nbuf][lo]);
            copy16(Wq   + gb + k0 + BK, &Bs[nbuf][lo]);
            __builtin_prefetch(Wq + gb + k0 + 2 * BK, 0, 1);  // global_prefetch_b8
        }

        // B fragments: two 16B chunks per lane (k kb..kb+7, kb+16..kb+23)
        union { v16bf v; v4f f[2]; } bfr[2];
#pragma unroll
        for (int tn = 0; tn < 2; ++tn) {
            const int n = waveN * 32 + tn * 16 + nrow;
            const v4f* p = (const v4f*)(&Bs[buf][n * BK + kb]);
            bfr[tn].f[0] = p[0];
            bfr[tn].f[1] = p[2];
        }

#pragma unroll
        for (int tm = 0; tm < 4; ++tm) {
            const int m = waveM * 64 + tm * 16 + arow;
            union { v16bf v; v4f f[2]; } ua, ur;
            ua.f[0] = *(const v4f*)(&As[buf][m * BK + ka]);
            ua.f[1] = *(const v4f*)(&As[buf][m * BK + ka + 16]);
            ur.f[0] = *(const v4f*)(&Ar[buf][m * BK + ka]);
            ur.f[1] = *(const v4f*)(&Ar[buf][m * BK + ka + 16]);
#pragma unroll
            for (int tn = 0; tn < 2; ++tn) {
                acc[tm][tn] = __builtin_amdgcn_wmma_f32_16x16x32_bf16(
                    false, ua.v, false, bfr[tn].v, (short)0, acc[tm][tn], false, false);
                acc[tm][tn] = __builtin_amdgcn_wmma_f32_16x16x32_bf16(
                    false, ur.v, false, bfr[tn].v, (short)0, acc[tm][tn], false, false);
            }
        }

        // next tile's copies must land, and all waves must pass, before swap
        wait_copies();
        __syncthreads();
    }

    // epilogue: C layout — lanes 0-15: M=i, N=lane; lanes 16-31: M=8+i, N=lane-16
#pragma unroll
    for (int tm = 0; tm < 4; ++tm) {
#pragma unroll
        for (int tn = 0; tn < 2; ++tn) {
            const int n = n0 + waveN * 32 + tn * 16 + (lane & 15);
            const int mb = m0 + waveM * 64 + tm * 16 + (lane >> 4) * 8;
            const float bn = bias[n];
#pragma unroll
            for (int i = 0; i < 8; ++i) {
                float v = acc[tm][tn][i] * scale + bn;
                out[(size_t)(mb + i) * NDIM + n] = __float2bfloat16(v);
            }
        }
    }
}

// ---------------------------------------------------------------------------
// Host-side launch
// ---------------------------------------------------------------------------
extern "C" void kernel_launch(void* const* d_in, const int* in_sizes, int n_in,
                              void* d_out, int out_size, void* d_ws, size_t ws_size,
                              hipStream_t stream) {
    const float* x    = (const float*)d_in[0];  // [8192,4096]
    const float* w    = (const float*)d_in[1];  // [4096,4096]
    const float* bias = (const float*)d_in[2];  // [4096]
    __hip_bfloat16* out = (__hip_bfloat16*)d_out;

    char* ws = (char*)d_ws;
    unsigned* hdr_u = (unsigned*)ws;
    float*    hdr_f = (float*)ws;
    bf16_t* wq  = (bf16_t*)(ws + 256);
    bf16_t* xsq = wq + (size_t)NDIM * KDIM;
    bf16_t* xrq = xsq + (size_t)MDIM * KDIM;

    init_hdr_kernel<<<1, 1, 0, stream>>>(hdr_u);
    maxabs_kernel<<<1024, 256, 0, stream>>>(x, (size_t)MDIM * KDIM, &hdr_u[0]);
    maxabs_kernel<<<1024, 256, 0, stream>>>(w, (size_t)NDIM * KDIM, &hdr_u[1]);
    scales_kernel<<<1, 1, 0, stream>>>(hdr_u, hdr_f);

    quantw_kernel<<<(NDIM * (size_t)KDIM / 32 + 255) / 256, 256, 0, stream>>>(w, hdr_f, wq);
    quantx_kernel<<<(MDIM * (size_t)KDIM / 32 + 255) / 256, 256, 0, stream>>>(x, hdr_f, xsq, xrq);

    dim3 grid(NDIM / BN, MDIM / BM);  // (32, 64)
    gemm_kernel<<<grid, 256, 0, stream>>>(xsq, xrq, wq, hdr_f, bias, out);
}
